// LSTM_47433618817626
// MI455X (gfx1250) — matrix operand order repaired and checked by reading, not compiled, not run on hardware
//
#include <hip/hip_runtime.h>
#include <hip/hip_bf16.h>

typedef __attribute__((ext_vector_type(16))) __bf16 v16bf;
typedef __attribute__((ext_vector_type(8)))  float  v8f;
typedef unsigned short ushort_t;
typedef unsigned int   uint_t;
typedef unsigned int   uint32x4 __attribute__((ext_vector_type(4)));
typedef int            int32x4  __attribute__((ext_vector_type(4)));
typedef int            int32x8  __attribute__((ext_vector_type(8)));

#define SEQ    256
#define HID    768
#define BATCH  256
#define NCLS   10
#define GATES  4
#define MROWS  (GATES*HID)        // 3072 stacked rows
#define KT     (HID/32)           // 24 K-tiles of 32
#define NT     (BATCH/16)         // 16 N-tiles of 16
#define MT     (MROWS/16)         // 192 M-tiles of 16
#define NWG    96                 // persistent workgroups (MT / MT_PER_WG)
#define MT_PER_WG 2
#define THREADS 256
#define TILE_ELEMS 512            // one 16x32 (or 32x16) bf16 fragment block
#define LDSW_ELEMS (MT_PER_WG * KT * TILE_ELEMS)   // 24576 bf16 = 48 KB

// ---- workspace layout (bytes) ----
static constexpr size_t BAR_OFF = 0;                                   // 2 x u32
static constexpr size_t WF_OFF  = 256;                                 // W frags
static constexpr size_t WF_BYTES  = (size_t)MT * KT * TILE_ELEMS * 2;  // 4,718,592
static constexpr size_t PRE_OFF = WF_OFF + WF_BYTES;
static constexpr size_t PRE_BYTES = (size_t)MROWS * BATCH * 4;         // 3,145,728
static constexpr size_t C_OFF   = PRE_OFF + PRE_BYTES;
static constexpr size_t C_BYTES   = (size_t)HID * BATCH * 4;           //   786,432
static constexpr size_t HB_OFF  = C_OFF + C_BYTES;
static constexpr size_t HB_BYTES  = (size_t)KT * NT * TILE_ELEMS * 2;  //   393,216
static constexpr size_t HL_OFF  = HB_OFF + HB_BYTES;

static __device__ __forceinline__ ushort_t f32_to_bf16(float f) {
    uint_t u = __float_as_uint(f);
    u += 0x7FFFu + ((u >> 16) & 1u);       // round-to-nearest-even
    return (ushort_t)(u >> 16);
}
static __device__ __forceinline__ float sigmoidf(float x) {
    return 1.0f / (1.0f + __expf(-x));
}

// ---- zero state + barrier ----
__global__ void lstm_init_kernel(float* c, ushort_t* hB, uint_t* bar) {
    int idx = blockIdx.x * blockDim.x + threadIdx.x;
    if (idx < HID * BATCH) { c[idx] = 0.0f; hB[idx] = 0; }
    if (idx < 2) bar[idx] = 0;
}

// ---- pack stacked W_*h (f32 row-major) -> bf16 WMMA A-fragment layout ----
// A tile 16(M)x32(K): lane = m + 16*(k/16), slot = k%16, per-lane v16bf contiguous.
__global__ void lstm_pack_kernel(const float* Wg, const float* Wi,
                                 const float* Wf, const float* Wo,
                                 ushort_t* WF) {
    int idx = blockIdx.x * blockDim.x + threadIdx.x;   // over MROWS*HID
    if (idx >= MROWS * HID) return;
    int r = idx / HID, k = idx % HID;
    int gate = r / HID;                                // 0..3 (g, i, f, o)
    int j = r - gate * HID;
    const float* W = (gate == 0) ? Wg : (gate == 1) ? Wi : (gate == 2) ? Wf : Wo;
    float v = W[j * HID + k];
    int mTile = r >> 4, m = r & 15;
    int kt = k >> 5,  kk = k & 31;
    int lane = m + 16 * (kk >> 4);
    int slot = kk & 15;
    WF[((size_t)(mTile * KT + kt)) * TILE_ELEMS + lane * 16 + slot] = f32_to_bf16(v);
}

// ---- grid-wide sense barrier ----
static __device__ __forceinline__ void grid_barrier(uint_t* cnt, uint_t* gen) {
    __syncthreads();
    if (threadIdx.x == 0) {
        __threadfence();
        uint_t g = __hip_atomic_load(gen, __ATOMIC_ACQUIRE, __HIP_MEMORY_SCOPE_AGENT);
        uint_t old = __hip_atomic_fetch_add(cnt, 1u, __ATOMIC_ACQ_REL, __HIP_MEMORY_SCOPE_AGENT);
        if (old == NWG - 1u) {
            __hip_atomic_store(cnt, 0u, __ATOMIC_RELAXED, __HIP_MEMORY_SCOPE_AGENT);
            __hip_atomic_fetch_add(gen, 1u, __ATOMIC_RELEASE, __HIP_MEMORY_SCOPE_AGENT);
        } else {
            while (__hip_atomic_load(gen, __ATOMIC_ACQUIRE, __HIP_MEMORY_SCOPE_AGENT) == g) {
                __builtin_amdgcn_s_sleep(2);
            }
        }
    }
    __syncthreads();
}

// ---- persistent recurrence kernel ----
__global__ void __launch_bounds__(THREADS)
lstm_persistent_kernel(const ushort_t* __restrict__ WF,
                       ushort_t* __restrict__ hB,
                       float* __restrict__ pre,
                       float* __restrict__ c,
                       float* __restrict__ hlin,
                       const float* __restrict__ x,
                       const float* __restrict__ Wgx, const float* __restrict__ Wix,
                       const float* __restrict__ Wfx, const float* __restrict__ Wox,
                       const float* __restrict__ bg,  const float* __restrict__ bi,
                       const float* __restrict__ bfv, const float* __restrict__ bo,
                       uint_t* bar) {
    __shared__ __align__(16) ushort_t ldsW[LDSW_ELEMS]; // 48 KB

    const int tid    = threadIdx.x;
    const int lane   = tid & 31;
    const int waveId = tid >> 5;                  // 0..7
    const int wg     = blockIdx.x;                // 0..95
    const int mBase  = wg * MT_PER_WG;            // first owned M-tile

    // ---- stage this WG's weight rows into LDS once (Tensor Data Mover) ----
#if defined(__AMDGCN__) && __has_builtin(__builtin_amdgcn_tensor_load_to_lds)
    if (waveId == 0) {
        unsigned long long ga =
            (unsigned long long)(WF + (size_t)mBase * KT * TILE_ELEMS);
        // Low 32 bits of the flat address of a __shared__ object are the LDS
        // byte offset (flat LDS aperture lives entirely in the high 32 bits).
        uint_t ldsOff = (uint_t)(unsigned long long)(size_t)&ldsW[0];
        // D# group 0: count=1 | lds_addr | global_addr | type=2
        uint32x4 g0;
        g0.x = 1u;
        g0.y = ldsOff;
        g0.z = (uint_t)(ga & 0xFFFFFFFFull);
        g0.w = (uint_t)((ga >> 32) & 0x1FFFFFFull) | (2u << 30);
        // D# group 1: data_size=2B; tensor_dim0 = tile_dim0 = 24576; dim1 = 1
        int32x8 g1;
        g1[0] = (1 << 16);                     // data_size code 1 (2 bytes)
        g1[1] = (int)((LDSW_ELEMS & 0xFFFF) << 16);   // tensor_dim0[15:0]
        g1[2] = (int)(((LDSW_ELEMS >> 16) & 0xFFFF) | (1u << 16)); // dim0 hi | tensor_dim1=1
        g1[3] = (int)((LDSW_ELEMS & 0xFFFF) << 16);   // tile_dim0
        g1[4] = 0;                             // tile_dim1/2 unused
        g1[5] = LDSW_ELEMS;                    // tensor_dim0_stride
        g1[6] = 0;
        g1[7] = 0;
        int32x4 gz4 = {0, 0, 0, 0};
#if defined(__clang_major__) && (__clang_major__ >= 23)
        int32x8 gz8 = {0, 0, 0, 0, 0, 0, 0, 0};
        __builtin_amdgcn_tensor_load_to_lds(g0, g1, gz4, gz4, gz8, 0);
#else
        __builtin_amdgcn_tensor_load_to_lds(g0, g1, gz4, gz4, 0);
#endif
        __builtin_amdgcn_s_wait_tensorcnt(0);
    }
#else
    {
        const uint4* src = (const uint4*)(WF + (size_t)mBase * KT * TILE_ELEMS);
        uint4*       dst = (uint4*)ldsW;
        for (int i = tid; i < LDSW_ELEMS / 8; i += THREADS)
            dst[i] = src[i];
    }
#endif
    __syncthreads();

    uint_t* bcnt = bar;
    uint_t* bgen = bar + 1;

    for (int t = 0; t < SEQ; ++t) {
        // ---- phase A: pre = W(3072x768, bf16 LDS) @ h(768x256, bf16 L2) ----
        // Each wave owns a 2x2 block of 16x16 tiles: A-frags shared across ni,
        // B-frags shared across mi -> 4 WMMAs per 4 fragment loads.
        {
            const int ni0 = waveId * 2;           // 0,2,..,14
            v8f acc00 = {}, acc01 = {}, acc10 = {}, acc11 = {};
            const ushort_t* lA0 = &ldsW[lane * 16];
            const ushort_t* lA1 = &ldsW[(size_t)KT * TILE_ELEMS + lane * 16];
            const ushort_t* gB0 = &hB[(size_t)ni0 * TILE_ELEMS + lane * 16];
            const ushort_t* gB1 = gB0 + TILE_ELEMS;
            #pragma unroll 2
            for (int kt = 0; kt < KT; ++kt) {
                size_t ao = (size_t)kt * TILE_ELEMS;
                size_t bo_ = (size_t)kt * NT * TILE_ELEMS;
                v16bf a0 = *(const v16bf*)(lA0 + ao);   // ds_load_b128 x2
                v16bf a1 = *(const v16bf*)(lA1 + ao);
                v16bf b0 = *(const v16bf*)(gB0 + bo_);  // global_load_b128 x2
                v16bf b1 = *(const v16bf*)(gB1 + bo_);
                if (kt + 1 < KT) {
                    __builtin_prefetch(gB0 + bo_ + (size_t)NT * TILE_ELEMS, 0, 1);
                    __builtin_prefetch(gB1 + bo_ + (size_t)NT * TILE_ELEMS, 0, 1);
                }
                acc00 = __builtin_amdgcn_wmma_f32_16x16x32_bf16(false, a0, false, b0, (short)0, acc00, false, false);
                acc01 = __builtin_amdgcn_wmma_f32_16x16x32_bf16(false, a0, false, b1, (short)0, acc01, false, false);
                acc10 = __builtin_amdgcn_wmma_f32_16x16x32_bf16(false, a1, false, b0, (short)0, acc10, false, false);
                acc11 = __builtin_amdgcn_wmma_f32_16x16x32_bf16(false, a1, false, b1, (short)0, acc11, false, false);
            }
            // C layout: VGPR r -> M = r (lanes 0-15) / r+8 (lanes 16-31); N = lane%16
            int colBase = ni0 * 16 + (lane & 15);
            int rowOff  = (lane < 16) ? 0 : 8;
            #pragma unroll
            for (int r = 0; r < 8; ++r) {
                int row0 = (mBase + 0) * 16 + rowOff + r;
                int row1 = (mBase + 1) * 16 + rowOff + r;
                pre[(size_t)row0 * BATCH + colBase]      = acc00[r];
                pre[(size_t)row0 * BATCH + colBase + 16] = acc01[r];
                pre[(size_t)row1 * BATCH + colBase]      = acc10[r];
                pre[(size_t)row1 * BATCH + colBase + 16] = acc11[r];
            }
        }
        grid_barrier(bcnt, bgen);

        // ---- phase B: gates + state update; write h back in B-fragment layout ----
        int gtid = wg * THREADS + tid;            // 0..24575
        #pragma unroll
        for (int p = 0; p < (HID * BATCH) / (NWG * THREADS); ++p) {
            int idx = gtid + p * (NWG * THREADS); // 0..196607
            int j = idx >> 8, b = idx & 255;
            float xv = x[b * SEQ + t];
            float pg = pre[(size_t)(0 * HID + j) * BATCH + b] + Wgx[j] * xv + bg[j];
            float pi = pre[(size_t)(1 * HID + j) * BATCH + b] + Wix[j] * xv + bi[j];
            float pf = pre[(size_t)(2 * HID + j) * BATCH + b] + Wfx[j] * xv + bfv[j];
            float po = pre[(size_t)(3 * HID + j) * BATCH + b] + Wox[j] * xv + bo[j];
            float g = tanhf(pg);
            float i = sigmoidf(pi);
            float f = sigmoidf(pf);
            float o = sigmoidf(po);
            float cn = g * i + c[idx] * f;
            c[idx] = cn;
            float hn = tanhf(cn) * o;
            int kt2 = j >> 5, ln2 = j & 31, nT = b >> 4, nn = b & 15;
            hB[(size_t)(kt2 * NT + nT) * TILE_ELEMS + (nn + 16 * (ln2 >> 4)) * 16 + (ln2 & 15)] = f32_to_bf16(hn);
            if (t == SEQ - 1) hlin[idx] = hn;     // keep h_T in f32 for projection
        }
        grid_barrier(bcnt, bgen);
    }
}

// ---- final projection: y = W_ph @ h_T + b_p  (10x256, trivial) ----
__global__ void lstm_proj_kernel(const float* __restrict__ hlin,
                                 const float* __restrict__ Wph,
                                 const float* __restrict__ bp,
                                 float* __restrict__ out) {
    int idx = blockIdx.x * blockDim.x + threadIdx.x;   // 0..2559
    if (idx >= NCLS * BATCH) return;
    int cls = idx / BATCH, b = idx % BATCH;
    float s = bp[cls];
    for (int k = 0; k < HID; ++k)
        s = fmaf(Wph[cls * HID + k], hlin[(size_t)k * BATCH + b], s);
    out[idx] = s;
}

extern "C" void kernel_launch(void* const* d_in, const int* in_sizes, int n_in,
                              void* d_out, int out_size, void* d_ws, size_t ws_size,
                              hipStream_t stream) {
    const float* x   = (const float*)d_in[0];
    const float* Wgx = (const float*)d_in[1];
    const float* Wgh = (const float*)d_in[2];
    const float* Wix = (const float*)d_in[3];
    const float* Wih = (const float*)d_in[4];
    const float* Wfx = (const float*)d_in[5];
    const float* Wfh = (const float*)d_in[6];
    const float* Wox = (const float*)d_in[7];
    const float* Woh = (const float*)d_in[8];
    const float* Wph = (const float*)d_in[9];
    const float* bg  = (const float*)d_in[10];
    const float* bi  = (const float*)d_in[11];
    const float* bfv = (const float*)d_in[12];
    const float* bo  = (const float*)d_in[13];
    const float* bp  = (const float*)d_in[14];

    char* ws = (char*)d_ws;
    uint_t*   bar  = (uint_t*)(ws + BAR_OFF);
    ushort_t* WF   = (ushort_t*)(ws + WF_OFF);
    float*    pre  = (float*)(ws + PRE_OFF);
    float*    c    = (float*)(ws + C_OFF);
    ushort_t* hB   = (ushort_t*)(ws + HB_OFF);
    float*    hlin = (float*)(ws + HL_OFF);

    lstm_init_kernel<<<(HID * BATCH + THREADS - 1) / THREADS, THREADS, 0, stream>>>(c, hB, bar);
    lstm_pack_kernel<<<(MROWS * HID) / THREADS, THREADS, 0, stream>>>(Wgh, Wih, Wfh, Woh, WF);
    lstm_persistent_kernel<<<NWG, THREADS, 0, stream>>>(WF, hB, pre, c, hlin, x,
                                                        Wgx, Wix, Wfx, Wox,
                                                        bg, bi, bfv, bo, bar);
    lstm_proj_kernel<<<(NCLS * BATCH + THREADS - 1) / THREADS, THREADS, 0, stream>>>(hlin, Wph, bp, (float*)d_out);
}
